// Transformer_79182017069172
// MI455X (gfx1250) — compile-verified, matrix operand order checked by reading, not run on hardware
//
#include <hip/hip_runtime.h>
#include <hip/hip_bf16.h>

// ---------------------------------------------------------------------------
// Transformer block for MI455X (gfx1250, wave32, WMMA f16 16x16x32, f32 acc)
// Round 2: 64x64 register-tiled GEMM (32.7 FLOP/B), dual-Q-tile attention.
// ---------------------------------------------------------------------------

typedef _Float16 f16;
typedef __attribute__((ext_vector_type(16))) _Float16 v16h;
typedef __attribute__((ext_vector_type(8)))  _Float16 v8h;
typedef __attribute__((ext_vector_type(8)))  float    v8f;

#define B_     8
#define N_     1025
#define D_     768
#define H_     12
#define F_     64
#define NPAD   1056            // 33 * 32, covers 1025 keys / 66 q-tiles
#define MROWS  (B_ * N_)       // 8200
#define MT64   129             // ceil(8200/64)
#define MPAD   (MT64 * 64)     // 8256
#define NCHUNK 96

// Load a 16x32 f16 operand fragment (A layout; B uses the same pattern with
// the 16-dim being the output column and K contiguous in memory).
// lane<16: rows 0..15, K {k..k+7, k+16..k+23}; lane>=16: K {k+8..k+15, k+24..k+31}
__device__ inline v16h load_frag16(const f16* base, int rstride) {
  int lane = threadIdx.x & 31;
  int r    = lane & 15;
  int ko   = (lane & 16) ? 8 : 0;
  const f16* p = base + (size_t)r * rstride + ko;
  v8h lo = *(const v8h*)(p);        // k = ko .. ko+7        (16B load)
  v8h hi = *(const v8h*)(p + 16);   // k = ko+16 .. ko+23    (16B load)
  v16h out;
#pragma unroll
  for (int i = 0; i < 8; ++i) { out[i] = lo[i]; out[i + 8] = hi[i]; }
  return out;
}

__device__ inline v8f wmma32(v16h a, v16h b, v8f c) {
  return __builtin_amdgcn_wmma_f32_16x16x32_f16(false, a, false, b, (short)0, c,
                                                false, false);
}

// ------------------------------ small kernels ------------------------------

__global__ __launch_bounds__(256) void cvt_kernel(const float* __restrict__ s,
                                                  f16* __restrict__ d, int n) {
  int i = blockIdx.x * 256 + threadIdx.x;
  if (i < n) d[i] = (f16)s[i];
}

__global__ __launch_bounds__(256) void ln_partial(const float* __restrict__ x,
                                                  float* __restrict__ part,
                                                  int perB, int nch) {
  int bidx = blockIdx.x;
  int b = bidx / nch, ck = bidx - b * nch;
  int chunk = perB / nch;                 // 787200 / 96 = 8200 exactly
  int start = ck * chunk;
  const float* xb = x + (size_t)b * perB;
  float s = 0.f, ss = 0.f;
  for (int i = start + threadIdx.x; i < start + chunk; i += 256) {
    float v = xb[i];
    s += v; ss += v * v;
  }
  __shared__ float r0[256], r1[256];
  r0[threadIdx.x] = s; r1[threadIdx.x] = ss;
  __syncthreads();
  for (int st = 128; st > 0; st >>= 1) {
    if (threadIdx.x < st) {
      r0[threadIdx.x] += r0[threadIdx.x + st];
      r1[threadIdx.x] += r1[threadIdx.x + st];
    }
    __syncthreads();
  }
  if (threadIdx.x == 0) {
    part[(size_t)bidx * 2]     = r0[0];
    part[(size_t)bidx * 2 + 1] = r1[0];
  }
}

__global__ __launch_bounds__(128) void ln_finish(const float* __restrict__ part,
                                                 float* __restrict__ stats,
                                                 int nch, float invn) {
  int b = blockIdx.x;
  float s = 0.f, ss = 0.f;
  for (int i = threadIdx.x; i < nch; i += 128) {
    s  += part[(size_t)(b * nch + i) * 2];
    ss += part[(size_t)(b * nch + i) * 2 + 1];
  }
  __shared__ float r0[128], r1[128];
  r0[threadIdx.x] = s; r1[threadIdx.x] = ss;
  __syncthreads();
  for (int st = 64; st > 0; st >>= 1) {
    if (threadIdx.x < st) {
      r0[threadIdx.x] += r0[threadIdx.x + st];
      r1[threadIdx.x] += r1[threadIdx.x + st];
    }
    __syncthreads();
  }
  if (threadIdx.x == 0) {
    float mu  = r0[0] * invn;
    float var = r1[0] * invn - mu * mu;
    stats[b * 2]     = mu;
    stats[b * 2 + 1] = rsqrtf(var + 1e-5f);
  }
}

__global__ __launch_bounds__(256) void ln_apply(const float* __restrict__ x,
                                                const float* __restrict__ w,
                                                const float* __restrict__ bsh,
                                                const float* __restrict__ stats,
                                                f16* __restrict__ y, int perB) {
  int i = blockIdx.x * 256 + threadIdx.x;   // grid sized exactly B_*perB
  int b = i / perB;
  int p = i - b * perB;
  float mu = stats[b * 2], rs = stats[b * 2 + 1];
  y[i] = (f16)((x[i] - mu) * rs * w[p] + bsh[p]);
}

// ------------------------------ WMMA GEMM ----------------------------------
// out[M,O] = A[M,K] @ W[O,K]^T ; one wave -> 64x64 tile (16 WMMA / 8KB loads);
// modes: 0: QKV scatter (q scaled 1/8, v transposed)  1: f32 = acc+bias+resid
//        2: f16 = gelu(acc+bias)
__global__ __launch_bounds__(256) void gemm_wmma(
    const f16* __restrict__ A, const f16* __restrict__ W,
    int K, int O, int mode,
    const float* __restrict__ bias, const float* __restrict__ resid,
    float* __restrict__ outf, f16* __restrict__ outh,
    f16* __restrict__ qb, f16* __restrict__ kb, f16* __restrict__ vtb) {
  int wave = threadIdx.x >> 5;
  int tile = blockIdx.x * 8 + wave;
  int ntiles = O >> 6;
  if (tile >= MT64 * ntiles) return;
  int nt = tile % ntiles;
  int mt = tile / ntiles;
  const f16* Ab = A + (size_t)mt * 64 * K;
  const f16* Wb = W + (size_t)nt * 64 * K;

  v8f acc[4][4] = {};
  for (int k0 = 0; k0 < K; k0 += 32) {
    v16h a0 = load_frag16(Ab + (size_t)0 * 16 * K + k0, K);
    v16h a1 = load_frag16(Ab + (size_t)1 * 16 * K + k0, K);
    v16h a2 = load_frag16(Ab + (size_t)2 * 16 * K + k0, K);
    v16h a3 = load_frag16(Ab + (size_t)3 * 16 * K + k0, K);
#pragma unroll
    for (int fc = 0; fc < 4; ++fc) {
      v16h b = load_frag16(Wb + (size_t)fc * 16 * K + k0, K);
      acc[0][fc] = wmma32(a0, b, acc[0][fc]);
      acc[1][fc] = wmma32(a1, b, acc[1][fc]);
      acc[2][fc] = wmma32(a2, b, acc[2][fc]);
      acc[3][fc] = wmma32(a3, b, acc[3][fc]);
    }
  }

  int lane = threadIdx.x & 31;
  int cl  = lane & 15;
  int rhi = (lane >> 4) << 3;              // C layout: row = j + 8*(lane>=16)
#pragma unroll
  for (int ms = 0; ms < 4; ++ms) {
    int rb = mt * 64 + ms * 16 + rhi;
#pragma unroll
    for (int fc = 0; fc < 4; ++fc) {
      int o = nt * 64 + fc * 16 + cl;
      if (mode == 0) {
        int three = o / D_;
        int rem = o - three * D_;
        int h = rem >> 6;
        int f = rem & 63;
#pragma unroll
        for (int j = 0; j < 8; ++j) {
          int m = rb + j;
          if (m >= MROWS) continue;
          int bb = m / N_;
          int n  = m - bb * N_;
          float val = acc[ms][fc][j];
          if (three == 0)
            qb[((size_t)(bb * H_ + h) * NPAD + n) * F_ + f] = (f16)(val * 0.125f);
          else if (three == 1)
            kb[((size_t)(bb * H_ + h) * NPAD + n) * F_ + f] = (f16)val;
          else
            vtb[((size_t)(bb * H_ + h) * F_ + f) * NPAD + n] = (f16)val;
        }
      } else if (mode == 1) {
        float bi = bias[o];
#pragma unroll
        for (int j = 0; j < 8; ++j) {
          int m = rb + j;
          if (m >= MROWS) continue;
          size_t idx = (size_t)m * D_ + o;
          outf[idx] = acc[ms][fc][j] + bi + resid[idx];
        }
      } else {
        float bi = bias[o];
#pragma unroll
        for (int j = 0; j < 8; ++j) {
          int m = rb + j;
          if (m >= MROWS) continue;
          float x = acc[ms][fc][j] + bi;
          outh[(size_t)m * D_ + o] = (f16)(0.5f * x * (1.0f + erff(x * 0.70710678118f)));
        }
      }
    }
  }
}

// ------------------------------ attention ----------------------------------
// One wave per (b,h, 32 query rows = two 16-row tiles). Flash softmax,
// 32 keys per step; K/V fragments shared between the two Q tiles.
__global__ __launch_bounds__(32) void attn_kernel(const f16* __restrict__ q,
                                                  const f16* __restrict__ k,
                                                  const f16* __restrict__ vt,
                                                  f16* __restrict__ o) {
  int qp = blockIdx.x;                  // 0..32 -> rows qp*32 .. qp*32+31
  int bh = blockIdx.y;
  int b = bh / H_, h = bh - b * H_;
  int lane = threadIdx.x;
  const f16* qbase = q  + ((size_t)bh * NPAD + qp * 32) * F_;
  const f16* kbase = k  + (size_t)bh * NPAD * F_;
  const f16* vbase = vt + (size_t)bh * F_ * NPAD;

  v16h qa[2][2];
  qa[0][0] = load_frag16(qbase, F_);
  qa[0][1] = load_frag16(qbase + 32, F_);
  qa[1][0] = load_frag16(qbase + (size_t)16 * F_, F_);
  qa[1][1] = load_frag16(qbase + (size_t)16 * F_ + 32, F_);

  v8f oa[2][4] = {};
  float mrow[2][8], lrow[2][8];
#pragma unroll
  for (int t = 0; t < 2; ++t)
#pragma unroll
    for (int j = 0; j < 8; ++j) { mrow[t][j] = -1e30f; lrow[t][j] = 0.f; }

  __shared__ __align__(16) f16 pbuf[2][16 * 32];

  int cl = lane & 15;
  int rhi = (lane >> 4) << 3;

  for (int j0 = 0; j0 < N_; j0 += 32) {
    v16h kb0 = load_frag16(kbase + (size_t)j0 * F_,              F_);
    v16h kb1 = load_frag16(kbase + (size_t)j0 * F_ + 32,         F_);
    v16h kb2 = load_frag16(kbase + (size_t)(j0 + 16) * F_,       F_);
    v16h kb3 = load_frag16(kbase + (size_t)(j0 + 16) * F_ + 32,  F_);

    bool v0 = (j0 + cl) < N_;
    bool v1 = (j0 + 16 + cl) < N_;
#pragma unroll
    for (int t = 0; t < 2; ++t) {
      v8f s0 = {}, s1 = {};
      s0 = wmma32(qa[t][0], kb0, s0);
      s0 = wmma32(qa[t][1], kb1, s0);
      s1 = wmma32(qa[t][0], kb2, s1);
      s1 = wmma32(qa[t][1], kb3, s1);
#pragma unroll
      for (int j = 0; j < 8; ++j) {
        float a0 = v0 ? (float)s0[j] : -1e30f;
        float a1 = v1 ? (float)s1[j] : -1e30f;
        float tt = fmaxf(a0, a1);
#pragma unroll
        for (int d = 1; d < 16; d <<= 1) tt = fmaxf(tt, __shfl_xor(tt, d, 32));
        float nm  = fmaxf(mrow[t][j], tt);
        float scl = __expf(mrow[t][j] - nm);
        mrow[t][j] = nm;
        float p0 = v0 ? __expf(a0 - nm) : 0.f;
        float p1 = v1 ? __expf(a1 - nm) : 0.f;
        float rs = p0 + p1;
#pragma unroll
        for (int d = 1; d < 16; d <<= 1) rs += __shfl_xor(rs, d, 32);
        lrow[t][j] = lrow[t][j] * scl + rs;
        int pr = rhi + j;
        pbuf[t][pr * 32 + cl]      = (f16)p0;
        pbuf[t][pr * 32 + 16 + cl] = (f16)p1;
        oa[t][0][j] *= scl; oa[t][1][j] *= scl;
        oa[t][2][j] *= scl; oa[t][3][j] *= scl;
      }
    }
    __syncthreads();
    v16h pa0 = load_frag16(pbuf[0], 32);          // P as A operand (16x32)
    v16h pa1 = load_frag16(pbuf[1], 32);
#pragma unroll
    for (int fc = 0; fc < 4; ++fc) {
      v16h vb = load_frag16(vbase + (size_t)fc * 16 * NPAD + j0, NPAD);
      oa[0][fc] = wmma32(pa0, vb, oa[0][fc]);
      oa[1][fc] = wmma32(pa1, vb, oa[1][fc]);
    }
    __syncthreads();
  }

#pragma unroll
  for (int t = 0; t < 2; ++t) {
#pragma unroll
    for (int j = 0; j < 8; ++j) {
      int n = qp * 32 + t * 16 + rhi + j;
      if (n >= N_) continue;
      float inv = 1.0f / lrow[t][j];
      size_t rowoff = ((size_t)b * N_ + n) * D_ + h * F_;
      o[rowoff + 0 * 16 + cl] = (f16)(oa[t][0][j] * inv);
      o[rowoff + 1 * 16 + cl] = (f16)(oa[t][1][j] * inv);
      o[rowoff + 2 * 16 + cl] = (f16)(oa[t][2][j] * inv);
      o[rowoff + 3 * 16 + cl] = (f16)(oa[t][3][j] * inv);
    }
  }
}

// ------------------------------ launcher -----------------------------------

extern "C" void kernel_launch(void* const* d_in, const int* in_sizes, int n_in,
                              void* d_out, int out_size, void* d_ws, size_t ws_size,
                              hipStream_t stream) {
  (void)in_sizes; (void)n_in; (void)out_size; (void)ws_size;
  const float* z     = (const float*)d_in[0];
  const float* ln1w  = (const float*)d_in[1];
  const float* ln1b  = (const float*)d_in[2];
  const float* qkvw  = (const float*)d_in[3];
  const float* projw = (const float*)d_in[4];
  const float* projb = (const float*)d_in[5];
  const float* ln2w  = (const float*)d_in[6];
  const float* ln2b  = (const float*)d_in[7];
  const float* w1    = (const float*)d_in[8];
  const float* b1    = (const float*)d_in[9];
  const float* w2    = (const float*)d_in[10];
  const float* b2    = (const float*)d_in[11];
  float* out = (float*)d_out;

  char* ws = (char*)d_ws;
  size_t off = 0;
  auto take = [&](size_t bytes) -> void* {
    void* p = ws + off;
    off += (bytes + 255) & ~(size_t)255;
    return p;
  };
  const size_t padded16 = (size_t)MPAD * D_ * sizeof(f16);
  const size_t qk16     = (size_t)B_ * H_ * NPAD * F_ * sizeof(f16);
  f16* z1h   = (f16*)take(padded16);
  f16* o16   = (f16*)take(padded16);
  f16* z2h   = (f16*)take(padded16);
  f16* h1h   = (f16*)take(padded16);
  f16* qb    = (f16*)take(qk16);
  f16* kb    = (f16*)take(qk16);
  f16* vtb   = (f16*)take(qk16);
  float* zA  = (float*)take((size_t)B_ * N_ * D_ * sizeof(float));
  f16* qkvwh = (f16*)take((size_t)3 * D_ * D_ * sizeof(f16));
  f16* projwh= (f16*)take((size_t)D_ * D_ * sizeof(f16));
  f16* w1h   = (f16*)take((size_t)D_ * D_ * sizeof(f16));
  f16* w2h   = (f16*)take((size_t)D_ * D_ * sizeof(f16));
  float* part  = (float*)take((size_t)B_ * NCHUNK * 2 * sizeof(float));
  float* stats = (float*)take((size_t)B_ * 2 * sizeof(float));

  // Zero padding regions (capture-safe async memsets).
  size_t tail = (size_t)(MPAD - MROWS) * D_ * sizeof(f16);
  hipMemsetAsync((char*)z1h + (size_t)MROWS * D_ * sizeof(f16), 0, tail, stream);
  hipMemsetAsync((char*)o16 + (size_t)MROWS * D_ * sizeof(f16), 0, tail, stream);
  hipMemsetAsync((char*)z2h + (size_t)MROWS * D_ * sizeof(f16), 0, tail, stream);
  hipMemsetAsync((char*)h1h + (size_t)MROWS * D_ * sizeof(f16), 0, tail, stream);
  hipMemsetAsync(qb, 0, qk16, stream);
  hipMemsetAsync(kb, 0, qk16, stream);
  hipMemsetAsync(vtb, 0, qk16, stream);

  // Weights -> f16
  int nqkv = 3 * D_ * D_;
  int nd2  = D_ * D_;
  cvt_kernel<<<(nqkv + 255) / 256, 256, 0, stream>>>(qkvw, qkvwh, nqkv);
  cvt_kernel<<<(nd2 + 255) / 256, 256, 0, stream>>>(projw, projwh, nd2);
  cvt_kernel<<<(nd2 + 255) / 256, 256, 0, stream>>>(w1, w1h, nd2);
  cvt_kernel<<<(nd2 + 255) / 256, 256, 0, stream>>>(w2, w2h, nd2);

  const int perB = N_ * D_;
  const int applyBlocks = (B_ * perB) / 256;   // 6297600 / 256 exactly

  // LN1
  ln_partial<<<B_ * NCHUNK, 256, 0, stream>>>(z, part, perB, NCHUNK);
  ln_finish<<<B_, 128, 0, stream>>>(part, stats, NCHUNK, 1.0f / perB);
  ln_apply<<<applyBlocks, 256, 0, stream>>>(z, ln1w, ln1b, stats, z1h, perB);

  // QKV projection (O = 2304)
  {
    int tiles = MT64 * ((3 * D_) >> 6);
    gemm_wmma<<<(tiles + 7) / 8, 256, 0, stream>>>(
        z1h, qkvwh, D_, 3 * D_, 0,
        nullptr, nullptr, nullptr, nullptr, qb, kb, vtb);
  }

  // Attention (33 dual-tiles cover 66*16 = 1056 = NPAD rows)
  attn_kernel<<<dim3(NPAD / 32, B_ * H_), 32, 0, stream>>>(qb, kb, vtb, o16);

  // Output projection + bias + residual -> zA (f32)
  {
    int tiles = MT64 * (D_ >> 6);
    gemm_wmma<<<(tiles + 7) / 8, 256, 0, stream>>>(
        o16, projwh, D_, D_, 1,
        projb, z, zA, nullptr, nullptr, nullptr, nullptr);
  }

  // LN2
  ln_partial<<<B_ * NCHUNK, 256, 0, stream>>>(zA, part, perB, NCHUNK);
  ln_finish<<<B_, 128, 0, stream>>>(part, stats, NCHUNK, 1.0f / perB);
  ln_apply<<<applyBlocks, 256, 0, stream>>>(zA, ln2w, ln2b, stats, z2h, perB);

  // MLP layer 1 + exact GELU -> h1h (f16)
  {
    int tiles = MT64 * (D_ >> 6);
    gemm_wmma<<<(tiles + 7) / 8, 256, 0, stream>>>(
        z2h, w1h, D_, D_, 2,
        b1, nullptr, nullptr, h1h, nullptr, nullptr, nullptr);
  }

  // MLP layer 2 + bias + residual -> out (f32)
  {
    int tiles = MT64 * (D_ >> 6);
    gemm_wmma<<<(tiles + 7) / 8, 256, 0, stream>>>(
        h1h, w2h, D_, D_, 1,
        b2, zA, out, nullptr, nullptr, nullptr, nullptr);
  }
}
